// _TemporalProbSparseBlock_36558761624259
// MI455X (gfx1250) — matrix operand order repaired and checked by reading, not compiled, not run on hardware
//
#include <hip/hip_runtime.h>
#include <math.h>

// ---------------- types / constants ----------------
typedef _Float16 v16h __attribute__((ext_vector_type(16)));
typedef float    v8f  __attribute__((ext_vector_type(8)));
typedef _Float16 h8   __attribute__((ext_vector_type(8)));

constexpr int Bc = 8, Sc = 2048, Dc = 512, Hc = 8, DHc = 64, Fc = 2048, TQMAX = 64;
constexpr float LN_EPS = 1e-5f;

__device__ __forceinline__ int compute_topq(const int* pf, const int* pmk) {
  int mk = *pmk; if (mk < 1) mk = 1;
  int tq = (int)ceilf((float)(*pf) * logf((float)Sc));
  if (tq < mk) tq = mk;
  if (tq > Sc) tq = Sc;
  return tq;
}

// A fragment (16x32 f16, M x K): lane<16 -> row=lane, K {0..7,16..23};
// lane>=16 -> row=lane-16, K {8..15,24..31}. Caller passes k0 = 0 or 8.
__device__ __forceinline__ v16h load_afrag(const _Float16* row, int k0) {
  h8 lo = *(const h8*)(row + k0);
  h8 hi = *(const h8*)(row + k0 + 16);
  v16h r;
#pragma unroll
  for (int i = 0; i < 8; ++i) { r[i] = lo[i]; r[8 + i] = hi[i]; }
  return r;
}

// B fragment (32x16 f16, K x N): lane<16 -> col=lane, K {0..7,16..23};
// lane>=16 -> col=lane-16, K {8..15,24..31}. Caller passes pointer offset by 0 or 8.
__device__ __forceinline__ v16h load_bfrag(const _Float16* p) {
  h8 lo = *(const h8*)(p);
  h8 hi = *(const h8*)(p + 16);
  v16h r;
#pragma unroll
  for (int i = 0; i < 8; ++i) { r[i] = lo[i]; r[8 + i] = hi[i]; }
  return r;
}

__device__ __forceinline__ float block_reduce_sum(float v, float* red) {
  int t = threadIdx.x;
  red[t] = v; __syncthreads();
  for (int k = 128; k > 0; k >>= 1) { if (t < k) red[t] += red[t + k]; __syncthreads(); }
  float r = red[0]; __syncthreads();
  return r;
}
__device__ __forceinline__ float block_reduce_max(float v, float* red) {
  int t = threadIdx.x;
  red[t] = v; __syncthreads();
  for (int k = 128; k > 0; k >>= 1) { if (t < k) red[t] = fmaxf(red[t], red[t + k]); __syncthreads(); }
  float r = red[0]; __syncthreads();
  return r;
}

// ---------------- fp32 -> f16 convert ----------------
__global__ void cvt_f16_kernel(const float* __restrict__ src, _Float16* __restrict__ dst, int n) {
  for (int i = blockIdx.x * blockDim.x + threadIdx.x; i < n; i += gridDim.x * blockDim.x)
    dst[i] = (_Float16)src[i];
}

// ---------------- LayerNorm (one row per block, D=512) ----------------
__global__ void layernorm_kernel(const float* __restrict__ x, const float* __restrict__ g,
                                 const float* __restrict__ b, float* __restrict__ y32,
                                 _Float16* __restrict__ y16) {
  __shared__ float red[256];
  const size_t row = blockIdx.x;
  const float* xr = x + row * Dc;
  int t = threadIdx.x;
  float s = 0.f;
  for (int d = t; d < Dc; d += 256) s += xr[d];
  float mean = block_reduce_sum(s, red) / (float)Dc;
  float v = 0.f;
  for (int d = t; d < Dc; d += 256) { float dd = xr[d] - mean; v += dd * dd; }
  float var = block_reduce_sum(v, red) / (float)Dc;
  float inv = rsqrtf(var + LN_EPS);
  for (int d = t; d < Dc; d += 256) {
    float o = (xr[d] - mean) * inv * g[d] + b[d];
    if (y32) y32[row * Dc + d] = o;
    if (y16) y16[row * Dc + d] = (_Float16)o;
  }
}

// ---------------- WMMA GEMM: C[M,N] = A[M,K] @ W[N,K]^T + bias ----------------
// block = 256 threads = 8 waves (4 M x 2 N), block tile 128M x 64N, K step 64.
// Each wave owns a 32x32 tile (2x2 WMMA accumulators) -> 8 WMMAs per barrier pair.
// Register double-buffering: next K-tile is loaded from global into VGPRs while
// the current tile's WMMAs execute, hiding HBM/L2 latency behind the matrix pipe.
template <bool GELU_, bool RESID_>
__global__ void gemm_wmma(const _Float16* __restrict__ A, const _Float16* __restrict__ W,
                          const float* __restrict__ bias, const float* __restrict__ Rsrc,
                          float* __restrict__ Cf, _Float16* __restrict__ Ch,
                          int M, int N, int K) {
  __shared__ _Float16 As[128][72];
  __shared__ _Float16 Ws[64][72];
  const int tid = threadIdx.x;
  const int w = tid >> 5, lane = tid & 31, lh = lane & 15;
  const int wm = w >> 1, wn = w & 1;               // 4 x 2 wave grid
  const int m0 = blockIdx.y * 128, n0 = blockIdx.x * 64;
  const int mB = wm * 32, nB = wn * 32;            // wave tile origin in block tile
  const int k0a = (lane < 16) ? 0 : 8;
  const int k0b = (lane < 16) ? 0 : 8;

  v8f c00, c01, c10, c11;
#pragma unroll
  for (int i = 0; i < 8; ++i) { c00[i] = 0.f; c01[i] = 0.f; c10[i] = 0.f; c11[i] = 0.f; }

  // staging coordinates
  const int ar = tid >> 1, ac = (tid & 1) * 32;    // A: 128 rows x 64 cols, 32 halves/thread
  const int wr = tid >> 2, wc = (tid & 3) * 16;    // W: 64 rows x 64 cols, 16 halves/thread
  const _Float16* Aptr = &A[(size_t)(m0 + ar) * K + ac];
  const _Float16* Wptr = &W[(size_t)(n0 + wr) * K + wc];

  // preload first K-tile into registers
  h8 ra[4], rw[2];
#pragma unroll
  for (int j = 0; j < 4; ++j) ra[j] = *(const h8*)(Aptr + j * 8);
#pragma unroll
  for (int j = 0; j < 2; ++j) rw[j] = *(const h8*)(Wptr + j * 8);

  for (int kt = 0; kt < K; kt += 64) {
    // commit current tile to LDS
#pragma unroll
    for (int j = 0; j < 4; ++j) *(h8*)&As[ar][ac + j * 8] = ra[j];
#pragma unroll
    for (int j = 0; j < 2; ++j) *(h8*)&Ws[wr][wc + j * 8] = rw[j];
    __syncthreads();
    // issue next tile's global loads early; latency overlaps the WMMAs below
    if (kt + 64 < K) {
#pragma unroll
      for (int j = 0; j < 4; ++j) ra[j] = *(const h8*)(Aptr + kt + 64 + j * 8);
#pragma unroll
      for (int j = 0; j < 2; ++j) rw[j] = *(const h8*)(Wptr + kt + 64 + j * 8);
    }
#pragma unroll
    for (int ks = 0; ks < 64; ks += 32) {
      v16h a0 = load_afrag(&As[mB + lh][ks], k0a);
      v16h a1 = load_afrag(&As[mB + 16 + lh][ks], k0a);
      v16h b0 = load_bfrag(&Ws[nB + lh][ks + k0b]);
      v16h b1 = load_bfrag(&Ws[nB + 16 + lh][ks + k0b]);
      c00 = __builtin_amdgcn_wmma_f32_16x16x32_f16(false, a0, false, b0, (short)0, c00, false, false);
      c01 = __builtin_amdgcn_wmma_f32_16x16x32_f16(false, a0, false, b1, (short)0, c01, false, false);
      c10 = __builtin_amdgcn_wmma_f32_16x16x32_f16(false, a1, false, b0, (short)0, c10, false, false);
      c11 = __builtin_amdgcn_wmma_f32_16x16x32_f16(false, a1, false, b1, (short)0, c11, false, false);
    }
    __syncthreads();
  }

  // epilogue
  const int lrow = (lane < 16) ? 0 : 8;
  const int ncol0 = n0 + nB + lh;
  const int ncol1 = ncol0 + 16;
  const float bv0 = bias ? bias[ncol0] : 0.f;
  const float bv1 = bias ? bias[ncol1] : 0.f;
  auto emit = [&](int m, int n, float v) {
    if (GELU_) v = 0.5f * v * (1.f + erff(v * 0.70710678118654752f));
    if (RESID_) v += Rsrc[(size_t)m * N + n];
    if (Cf) Cf[(size_t)m * N + n] = v;
    if (Ch) Ch[(size_t)m * N + n] = (_Float16)v;
  };
#pragma unroll
  for (int r = 0; r < 8; ++r) {
    const int ma = m0 + mB + r + lrow;
    const int mb = ma + 16;
    emit(ma, ncol0, c00[r] + bv0);
    emit(ma, ncol1, c01[r] + bv1);
    emit(mb, ncol0, c10[r] + bv0);
    emit(mb, ncol1, c11[r] + bv1);
  }
}

// ---------------- ProbSparse score stats (WMMA): sparsity = rowmax - rowmean ----------------
// grid (Sc/16, Bc); each block: 16 queries vs all 2048 keys; 8 waves stride key tiles.
__global__ void score_stats_kernel(const _Float16* __restrict__ xn16, float* __restrict__ sparsity) {
  __shared__ _Float16 Qs[16][520];
  __shared__ float mbuf[16][128];
  __shared__ float sbuf[16][128];
  const int b = blockIdx.y, q0 = blockIdx.x * 16;
  const int tid = threadIdx.x, w = tid >> 5, lane = tid & 31, lh = lane & 15;
  const _Float16* xb = xn16 + (size_t)b * Sc * Dc;

  { // stage Q tile 16 x 512
    int r = tid >> 4, c0 = (tid & 15) * 32;
#pragma unroll
    for (int j = 0; j < 32; j += 8)
      *(h8*)&Qs[r][c0 + j] = *(const h8*)&xb[(size_t)(q0 + r) * Dc + c0 + j];
  }
  __syncthreads();

  const int k0a = (lane < 16) ? 0 : 8;
  const int k0b = (lane < 16) ? 0 : 8;
  const float scale = 0.044194173824159216f; // 1/sqrt(512)
  float lmax[8], lsum[8];
#pragma unroll
  for (int r = 0; r < 8; ++r) { lmax[r] = -1e30f; lsum[r] = 0.f; }

  for (int kt = w; kt < Sc / 16; kt += 8) {
    v8f c;
#pragma unroll
    for (int i = 0; i < 8; ++i) c[i] = 0.f;
    const _Float16* krow = xb + (size_t)(kt * 16 + lh) * Dc + k0b;
#pragma unroll 4
    for (int ks = 0; ks < Dc; ks += 32) {
      v16h a = load_afrag(&Qs[lh][ks], k0a);
      v16h bb = load_bfrag(krow + ks);
      c = __builtin_amdgcn_wmma_f32_16x16x32_f16(false, a, false, bb, (short)0, c, false, false);
    }
#pragma unroll
    for (int r = 0; r < 8; ++r) {
      float v = c[r] * scale;
      lmax[r] = fmaxf(lmax[r], v);
      lsum[r] += v;
    }
  }
  const int rb = (lane < 16) ? 0 : 8;
#pragma unroll
  for (int r = 0; r < 8; ++r) { mbuf[rb + r][w * 16 + lh] = lmax[r]; sbuf[rb + r][w * 16 + lh] = lsum[r]; }
  __syncthreads();
  if (tid < 16) {
    float m = -1e30f, s = 0.f;
    for (int j = 0; j < 128; ++j) { m = fmaxf(m, mbuf[tid][j]); s += sbuf[tid][j]; }
    sparsity[(size_t)b * Sc + q0 + tid] = m - s / (float)Sc;
  }
}

// ---------------- top-k (iterative block argmax in LDS) ----------------
__global__ void topk_kernel(const float* __restrict__ sparsity, int* __restrict__ top_idx,
                            const int* pf, const int* pmk) {
  __shared__ float sv[Sc];
  __shared__ float rv[256];
  __shared__ int ri[256];
  const int b = blockIdx.x, t = threadIdx.x;
  for (int i = t; i < Sc; i += 256) sv[i] = sparsity[(size_t)b * Sc + i];
  __syncthreads();
  const int tq = compute_topq(pf, pmk);
  for (int round = 0; round < tq; ++round) {
    float bv = -1e30f; int bi = 0;
    for (int i = t; i < Sc; i += 256) { float v = sv[i]; if (v > bv) { bv = v; bi = i; } }
    rv[t] = bv; ri[t] = bi; __syncthreads();
    for (int k = 128; k > 0; k >>= 1) {
      if (t < k && rv[t + k] > rv[t]) { rv[t] = rv[t + k]; ri[t] = ri[t + k]; }
      __syncthreads();
    }
    if (t == 0) { top_idx[b * TQMAX + round] = ri[0]; sv[ri[0]] = -1e30f; }
    __syncthreads();
  }
  if (t == 0) for (int r = tq; r < TQMAX; ++r) top_idx[b * TQMAX + r] = 0;
}

// ---------------- gather sparse queries (f16) ----------------
__global__ void gather_q_kernel(const _Float16* __restrict__ xn16, const int* __restrict__ top_idx,
                                _Float16* __restrict__ sq16, const int* pf, const int* pmk) {
  const int i = blockIdx.x, b = blockIdx.y, t = threadIdx.x;
  const int tq = compute_topq(pf, pmk);
  const size_t dst = (size_t)(b * TQMAX + i) * Dc;
  if (i < tq) {
    const _Float16* src = xn16 + ((size_t)b * Sc + top_idx[b * TQMAX + i]) * Dc;
    for (int d = t; d < Dc; d += 256) sq16[dst + d] = src[d];
  } else {
    for (int d = t; d < Dc; d += 256) sq16[dst + d] = (_Float16)0.f;
  }
}

// ---------------- sparse softmax attention (39 queries; VALU path, ~1.5% of FLOPs) ----------------
__global__ void sparse_attn_kernel(const float* __restrict__ qproj, const _Float16* __restrict__ k16,
                                   const _Float16* __restrict__ v16, _Float16* __restrict__ sctx16,
                                   const int* pf, const int* pmk) {
  const int qi = blockIdx.x, h = blockIdx.y, b = blockIdx.z, t = threadIdx.x;
  const int tq = compute_topq(pf, pmk);
  __shared__ float qv[DHc];
  __shared__ float sc[Sc];
  __shared__ float red[256];
  const size_t qrow = (size_t)(b * TQMAX + qi) * Dc + h * DHc;
  if (qi >= tq) {
    if (h == 0) for (int d = t; d < Dc; d += 256) sctx16[(size_t)(b * TQMAX + qi) * Dc + d] = (_Float16)0.f;
    return;
  }
  if (t < DHc) qv[t] = qproj[qrow + t];
  __syncthreads();
  // scores
  float lmax = -1e30f;
  for (int j = 0; j < 8; ++j) {
    int key = t * 8 + j;
    const _Float16* kr = k16 + ((size_t)b * Sc + key) * Dc + h * DHc;
    float acc = 0.f;
#pragma unroll 16
    for (int d = 0; d < DHc; ++d) acc += qv[d] * (float)kr[d];
    acc *= 0.125f; // 1/sqrt(64)
    sc[key] = acc;
    lmax = fmaxf(lmax, acc);
  }
  __syncthreads();
  float mx = block_reduce_max(lmax, red);
  float lsum = 0.f;
  for (int j = 0; j < 8; ++j) {
    int key = t * 8 + j;
    float p = __expf(sc[key] - mx);
    sc[key] = p;
    lsum += p;
  }
  __syncthreads();
  float smv = block_reduce_sum(lsum, red);
  // ctx = probs @ V
  const int d = t & 63, chunk = t >> 6;
  float acc = 0.f;
  for (int key = chunk * 512; key < chunk * 512 + 512; ++key)
    acc += sc[key] * (float)v16[((size_t)b * Sc + key) * Dc + h * DHc + d];
  red[t] = acc; __syncthreads();
  if (t < DHc) {
    float ctx = (red[t] + red[t + 64] + red[t + 128] + red[t + 192]) / smv;
    sctx16[(size_t)(b * TQMAX + qi) * Dc + h * DHc + t] = (_Float16)ctx;
  }
}

// ---------------- base = mean over S of x_norm ----------------
__global__ void base_mean_kernel(const float* __restrict__ xn32, float* __restrict__ base) {
  int id = blockIdx.x * 256 + threadIdx.x;
  if (id >= Bc * Dc) return;
  int b = id / Dc, d = id % Dc;
  const float* p = xn32 + (size_t)b * Sc * Dc + d;
  float s = 0.f;
  for (int i = 0; i < Sc; ++i) s += p[(size_t)i * Dc];
  base[id] = s / (float)Sc;
}

// ---------------- x2 = x + broadcast(base) ----------------
__global__ void residual_base_kernel(const float* __restrict__ x, const float* __restrict__ base,
                                     float* __restrict__ x2) {
  const size_t total = (size_t)Bc * Sc * Dc;
  for (size_t i = blockIdx.x * 256ull + threadIdx.x; i < total; i += (size_t)gridDim.x * 256ull) {
    int d = (int)(i % Dc);
    int b = (int)(i / ((size_t)Sc * Dc));
    x2[i] = x[i] + base[b * Dc + d];
  }
}

// ---------------- scatter sparse attention rows: x2[b,idx,:] = x[b,idx,:] + sparse_out ----------------
__global__ void scatter_kernel(const float* __restrict__ x, const int* __restrict__ top_idx,
                               const float* __restrict__ sout, float* __restrict__ x2,
                               const int* pf, const int* pmk) {
  const int i = blockIdx.x, b = blockIdx.y, t = threadIdx.x;
  const int tq = compute_topq(pf, pmk);
  if (i >= tq) return;
  const int row = top_idx[b * TQMAX + i];
  const size_t dst = ((size_t)b * Sc + row) * Dc;
  const size_t src = (size_t)(b * TQMAX + i) * Dc;
  for (int d = t; d < Dc; d += 256) x2[dst + d] = x[dst + d] + sout[src + d];
}

// ---------------- ratio scalar ----------------
__global__ void ratio_kernel(float* __restrict__ out, const int* pf, const int* pmk) {
  if (threadIdx.x == 0 && blockIdx.x == 0) {
    int tq = compute_topq(pf, pmk);
    out[(size_t)Bc * Sc * Dc] = (float)tq / (float)Sc;
  }
}

// ---------------- launch ----------------
extern "C" void kernel_launch(void* const* d_in, const int* in_sizes, int n_in,
                              void* d_out, int out_size, void* d_ws, size_t ws_size,
                              hipStream_t stream) {
  (void)in_sizes; (void)n_in; (void)out_size; (void)ws_size;
  const float* x     = (const float*)d_in[0];
  const float* ln1_g = (const float*)d_in[1];
  const float* ln1_b = (const float*)d_in[2];
  const float* in_w  = (const float*)d_in[3];
  const float* in_b  = (const float*)d_in[4];
  const float* out_w = (const float*)d_in[5];
  const float* out_b = (const float*)d_in[6];
  const float* ln2_g = (const float*)d_in[7];
  const float* ln2_b = (const float*)d_in[8];
  const float* w1    = (const float*)d_in[9];
  const float* b1    = (const float*)d_in[10];
  const float* w2    = (const float*)d_in[11];
  const float* b2    = (const float*)d_in[12];
  const int*   pf    = (const int*)d_in[13];
  const int*   pmk   = (const int*)d_in[14];
  float* out = (float*)d_out;

  const size_t NBSD = (size_t)Bc * Sc * Dc;
  char* wsb = (char*)d_ws;
  size_t off = 0;
  auto alloc = [&](size_t bytes) -> void* {
    size_t o = (off + 255) & ~(size_t)255;
    off = o + bytes;
    return (void*)(wsb + o);
  };

  // Phase-1 pool (dead before FFN): xn32 | xn16 | k16  == exactly 67,108,864 B == ffn1 size
  float*    xn32  = (float*)alloc(NBSD * 4);
  _Float16* xn16  = (_Float16*)alloc(NBSD * 2);
  _Float16* k16   = (_Float16*)alloc(NBSD * 2);
  _Float16* v16p  = (_Float16*)alloc(NBSD * 2);
  _Float16* ffn1  = (_Float16*)xn32; // alias: B*S*F f16 over [xn32, xn16, k16]
  float*    x2    = (float*)alloc(NBSD * 4);
  _Float16* h16   = (_Float16*)alloc(NBSD * 2);
  float*    spars = (float*)alloc((size_t)Bc * Sc * 4);
  int*      tidx  = (int*)alloc((size_t)Bc * TQMAX * 4);
  _Float16* sq16  = (_Float16*)alloc((size_t)Bc * TQMAX * Dc * 2);
  float*    qproj = (float*)alloc((size_t)Bc * TQMAX * Dc * 4);
  _Float16* sctx  = (_Float16*)alloc((size_t)Bc * TQMAX * Dc * 2);
  float*    sout  = (float*)alloc((size_t)Bc * TQMAX * Dc * 4);
  float*    base  = (float*)alloc((size_t)Bc * Dc * 4);
  _Float16* inw16 = (_Float16*)alloc((size_t)3 * Dc * Dc * 2);
  _Float16* outw16= (_Float16*)alloc((size_t)Dc * Dc * 2);
  _Float16* w1_16 = (_Float16*)alloc((size_t)Fc * Dc * 2);
  _Float16* w2_16 = (_Float16*)alloc((size_t)Dc * Fc * 2);

  // 0) weights -> f16
  cvt_f16_kernel<<<512, 256, 0, stream>>>(in_w,  inw16,  3 * Dc * Dc);
  cvt_f16_kernel<<<256, 256, 0, stream>>>(out_w, outw16, Dc * Dc);
  cvt_f16_kernel<<<512, 256, 0, stream>>>(w1,    w1_16,  Fc * Dc);
  cvt_f16_kernel<<<512, 256, 0, stream>>>(w2,    w2_16,  Dc * Fc);

  // 1) LN1
  layernorm_kernel<<<Bc * Sc, 256, 0, stream>>>(x, ln1_g, ln1_b, xn32, xn16);

  // 2) ProbSparse score stats (WMMA) + top-k + gather
  score_stats_kernel<<<dim3(Sc / 16, Bc), 256, 0, stream>>>(xn16, spars);
  topk_kernel<<<Bc, 256, 0, stream>>>(spars, tidx, pf, pmk);
  gather_q_kernel<<<dim3(TQMAX, Bc), 256, 0, stream>>>(xn16, tidx, sq16, pf, pmk);

  // 3) projections (WMMA GEMMs; block tile 128x64, double-buffered)
  gemm_wmma<false, false><<<dim3(Dc / 64, (Bc * Sc) / 128), 256, 0, stream>>>(
      xn16, inw16 + (size_t)Dc * Dc, in_b + Dc, nullptr, nullptr, k16, Bc * Sc, Dc, Dc);
  gemm_wmma<false, false><<<dim3(Dc / 64, (Bc * Sc) / 128), 256, 0, stream>>>(
      xn16, inw16 + (size_t)2 * Dc * Dc, in_b + 2 * Dc, nullptr, nullptr, v16p, Bc * Sc, Dc, Dc);
  gemm_wmma<false, false><<<dim3(Dc / 64, (Bc * TQMAX) / 128), 256, 0, stream>>>(
      sq16, inw16, in_b, nullptr, qproj, nullptr, Bc * TQMAX, Dc, Dc);

  // 4) sparse attention + output projection
  sparse_attn_kernel<<<dim3(TQMAX, Hc, Bc), 256, 0, stream>>>(qproj, k16, v16p, sctx, pf, pmk);
  gemm_wmma<false, false><<<dim3(Dc / 64, (Bc * TQMAX) / 128), 256, 0, stream>>>(
      sctx, outw16, out_b, nullptr, sout, nullptr, Bc * TQMAX, Dc, Dc);

  // 5) attn_out assembly + residual
  base_mean_kernel<<<(Bc * Dc + 255) / 256, 256, 0, stream>>>(xn32, base);
  residual_base_kernel<<<4096, 256, 0, stream>>>(x, base, x2);
  scatter_kernel<<<dim3(TQMAX, Bc), 256, 0, stream>>>(x, tidx, sout, x2, pf, pmk);

  // 6) LN2 + FFN (WMMA, GELU fused; FFN2 fuses bias + residual -> d_out)
  layernorm_kernel<<<Bc * Sc, 256, 0, stream>>>(x2, ln2_g, ln2_b, nullptr, h16);
  gemm_wmma<true, false><<<dim3(Fc / 64, (Bc * Sc) / 128), 256, 0, stream>>>(
      h16, w1_16, b1, nullptr, nullptr, ffn1, Bc * Sc, Fc, Dc);
  gemm_wmma<false, true><<<dim3(Dc / 64, (Bc * Sc) / 128), 256, 0, stream>>>(
      ffn1, w2_16, b2, x2, out, nullptr, Bc * Sc, Dc, Fc);

  // 7) ratio scalar
  ratio_kernel<<<1, 1, 0, stream>>>(out, pf, pmk);
}
